// SlayerMLP_59407987638664
// MI455X (gfx1250) — compile-verified, matrix operand order checked by reading, not run on hardware
//
#include <hip/hip_runtime.h>

// ---------------------------------------------------------------------------
// Fully fused SLAYER 2-layer SNN for MI455X (gfx1250, wave32).
// One wave (32 lanes) per batch sample; time tiled by 32.
//   GEMMs      -> v_wmma_f32_16x16x32_bf16 (spikes are exactly 0/1 in bf16)
//   alpha conv -> 2-state IIR (exact generator of (t/tau)e^{1-t/tau})
//   refractory -> same IIR trick (tau=1), folded into the threshold scan
// All intermediates live in LDS/VGPRs: HBM traffic = read x once + write s2.
// ---------------------------------------------------------------------------

typedef __attribute__((ext_vector_type(16))) __bf16  v16bf;
typedef __attribute__((ext_vector_type(8)))  float   v8f;
typedef __attribute__((ext_vector_type(4)))  unsigned int v4u;

union BFrag {
    v16bf          v;
    v4u            q[2];
    unsigned short h[16];
};

__device__ __forceinline__ unsigned short f2bf(float f) {
    unsigned int u = __float_as_uint(f);
    unsigned int r = u + 0x7FFFu + ((u >> 16) & 1u);   // round-to-nearest-even
    return (unsigned short)(r >> 16);
}

// SRM psp IIR:   k[s] = (s/10) e^{1-s/10} = (e/10) * s * d^s,  d = e^{-0.1}
// refractory:    k[s] = -20 s e^{1-s}     = -20e   * s * dr^s, dr = e^{-1}
__constant__ constexpr float KD1 = 0.9048374180359596f;   // exp(-1/10)
__constant__ constexpr float KC1 = 0.2718281828459045f;   // Ts * e / tau_sr
__constant__ constexpr float KDR = 0.36787944117144233f;  // exp(-1)
__constant__ constexpr float KCR = -54.365636569180902f;  // -scale*theta*e = -20e
__constant__ constexpr float THETA = 10.0f;

__global__ __launch_bounds__(32)
void slayer_fused(const float* __restrict__ X,   // [N,156,325]
                  const float* __restrict__ W1,  // [32,156]
                  const float* __restrict__ W2,  // [20,32]
                  float* __restrict__ Out)       // [N,20,325]
{
    constexpr int CIN = 156, T = 325, C2 = 20, TT = 32;
    constexpr int NT  = (T + TT - 1) / TT;       // 11 time tiles
    constexpr int XS  = 168;                     // xt row stride (halves), 16B-aligned rows
    constexpr int S1S = 40;                      // s1t row stride (halves)
    constexpr int ZS  = 33;                      // zt row stride (floats), bank-conflict-free

    __shared__ __align__(16) unsigned short xt [TT][XS];   // input tile, bf16, [time][K]
    __shared__ __align__(16) unsigned short s1t[TT][S1S];  // layer-1 spikes, bf16, [time][K]
    __shared__ float zt [32][ZS];                          // psp pre-activation [chan][time]
    __shared__ float s2t[C2][TT];                          // layer-2 spikes staging

    const int lane = threadIdx.x;
    const int lg   = lane >> 4;      // lane group (K-half select in WMMA operands)
    const int lr   = lane & 15;      // M (A) / N (B,D) index within a 16x16 tile
    const int n    = blockIdx.x;

    const float* Xn = X   + (size_t)n * CIN * T;
    float*       On = Out + (size_t)n * C2  * T;

    // ---- preload W1 / W2 WMMA A-fragments into VGPRs (bf16) -------------
    // 16-bit A layout: lane holds row M=lr; elems 0..7 -> K = base+8*lg+e,
    // elems 8..15 -> K = base+16+8*lg+e  (base = 32*kb).
    BFrag af1[2][5];
    #pragma unroll
    for (int mt = 0; mt < 2; ++mt) {
        const int m = mt * 16 + lr;
        #pragma unroll
        for (int kb = 0; kb < 5; ++kb) {
            #pragma unroll
            for (int e = 0; e < 8; ++e) {
                const int k0 = kb * 32 + lg * 8 + e;
                const int k1 = kb * 32 + 16 + lg * 8 + e;
                af1[mt][kb].h[e]     = (k0 < CIN) ? f2bf(W1[m * CIN + k0]) : (unsigned short)0;
                af1[mt][kb].h[8 + e] = (k1 < CIN) ? f2bf(W1[m * CIN + k1]) : (unsigned short)0;
            }
        }
    }
    BFrag af2[2];
    #pragma unroll
    for (int mt = 0; mt < 2; ++mt) {
        const int m  = mt * 16 + lr;
        const bool ok = (m < C2);
        #pragma unroll
        for (int e = 0; e < 8; ++e) {
            af2[mt].h[e]     = ok ? f2bf(W2[m * 32 + lg * 8 + e])      : (unsigned short)0;
            af2[mt].h[8 + e] = ok ? f2bf(W2[m * 32 + 16 + lg * 8 + e]) : (unsigned short)0;
        }
    }

    const v8f zero8 = {0.f, 0.f, 0.f, 0.f, 0.f, 0.f, 0.f, 0.f};

    // per-channel streaming state (psp IIR + refractory IIR), layer 1 & 2
    float a1 = 0.f, b1 = 0.f, ar1 = 0.f, br1 = 0.f;
    float a2 = 0.f, b2 = 0.f, ar2 = 0.f, br2 = 0.f;

    for (int tile = 0; tile < NT; ++tile) {
        const int t0  = tile * TT;
        const int t   = t0 + lane;
        const bool tin = (t < T);
        const int ttv = (T - t0 < TT) ? (T - t0) : TT;

        // ---- stage input tile into LDS as bf16, [time][K] ---------------
        #pragma unroll 4
        for (int i = 0; i < CIN; ++i) {
            const float v = tin ? Xn[i * T + t] : 0.0f;   // coalesced over lanes
            xt[lane][i] = f2bf(v);
        }
        #pragma unroll
        for (int i = CIN; i < 160; ++i) xt[lane][i] = 0;  // K zero-padding (156->160)

        if (tile + 1 < NT) {                               // prefetch next tile
            for (int i = lane; i < CIN; i += 32)
                __builtin_prefetch(&Xn[i * T + t0 + TT], 0, 1);
        }
        __syncthreads();

        // ---- layer-1 GEMM: z1(32x32) = W1(32x160) x xt(160x32) ----------
        v8f acc[2][2];
        #pragma unroll
        for (int mt = 0; mt < 2; ++mt)
            #pragma unroll
            for (int nt = 0; nt < 2; ++nt) acc[mt][nt] = zero8;

        #pragma unroll
        for (int nt = 0; nt < 2; ++nt) {
            const unsigned short* xrow = &xt[nt * 16 + lr][0];
            #pragma unroll
            for (int kb = 0; kb < 5; ++kb) {
                BFrag bf;                                   // B-frag: 2x ds_load_b128
                bf.q[0] = *(const v4u*)(xrow + kb * 32 + lg * 8);
                bf.q[1] = *(const v4u*)(xrow + kb * 32 + 16 + lg * 8);
                #pragma unroll
                for (int mt = 0; mt < 2; ++mt)
                    acc[mt][nt] = __builtin_amdgcn_wmma_f32_16x16x32_bf16(
                        false, af1[mt][kb].v, false, bf.v, (short)0,
                        acc[mt][nt], false, false);
            }
        }
        // D layout: VGPR j -> M = j + 8*lg, N = lr  ->  zt[channel][time]
        #pragma unroll
        for (int mt = 0; mt < 2; ++mt)
            #pragma unroll
            for (int nt = 0; nt < 2; ++nt)
                #pragma unroll
                for (int j = 0; j < 8; ++j)
                    zt[mt * 16 + lg * 8 + j][nt * 16 + lr] = acc[mt][nt][j];
        __syncthreads();

        // ---- layer-1 scan: lane == channel (exactly 32) -----------------
        for (int tt = 0; tt < ttv; ++tt) {
            const float z = zt[lane][tt];
            b1 = KD1 * (b1 + a1);            // psp IIR (uses a[t-1])
            a1 = KD1 * a1 + z;
            br1 = KDR * (br1 + ar1);         // refractory IIR (spikes up to t-1)
            const float u = KC1 * b1 + KCR * br1;
            const float s = (u >= THETA) ? 1.0f : 0.0f;
            ar1 = KDR * ar1 + s;
            s1t[tt][lane] = (u >= THETA) ? (unsigned short)0x3F80 : (unsigned short)0;
        }
        __syncthreads();

        // ---- layer-2 GEMM: z2(32x32) = W2(32pad x 32) x s1t(32x32) ------
        v8f acc2[2][2];
        #pragma unroll
        for (int mt = 0; mt < 2; ++mt)
            #pragma unroll
            for (int nt = 0; nt < 2; ++nt) acc2[mt][nt] = zero8;

        #pragma unroll
        for (int nt = 0; nt < 2; ++nt) {
            const unsigned short* srow = &s1t[nt * 16 + lr][0];
            BFrag bf;
            bf.q[0] = *(const v4u*)(srow + lg * 8);
            bf.q[1] = *(const v4u*)(srow + 16 + lg * 8);
            #pragma unroll
            for (int mt = 0; mt < 2; ++mt)
                acc2[mt][nt] = __builtin_amdgcn_wmma_f32_16x16x32_bf16(
                    false, af2[mt].v, false, bf.v, (short)0,
                    acc2[mt][nt], false, false);
        }
        #pragma unroll
        for (int mt = 0; mt < 2; ++mt)
            #pragma unroll
            for (int nt = 0; nt < 2; ++nt)
                #pragma unroll
                for (int j = 0; j < 8; ++j)
                    zt[mt * 16 + lg * 8 + j][nt * 16 + lr] = acc2[mt][nt][j];
        __syncthreads();

        // ---- layer-2 scan (channels 0..19) ------------------------------
        if (lane < C2) {
            for (int tt = 0; tt < ttv; ++tt) {
                const float z = zt[lane][tt];
                b2 = KD1 * (b2 + a2);
                a2 = KD1 * a2 + z;
                br2 = KDR * (br2 + ar2);
                const float u = KC1 * b2 + KCR * br2;
                const float s = (u >= THETA) ? 1.0f : 0.0f;
                ar2 = KDR * ar2 + s;
                s2t[lane][tt] = s;
            }
        }
        __syncthreads();

        // ---- coalesced output stores ------------------------------------
        if (tin) {
            #pragma unroll 4
            for (int c = 0; c < C2; ++c)
                On[c * T + t] = s2t[c][lane];
        }
        __syncthreads();
    }
}

extern "C" void kernel_launch(void* const* d_in, const int* in_sizes, int n_in,
                              void* d_out, int out_size, void* d_ws, size_t ws_size,
                              hipStream_t stream) {
    const float* X  = (const float*)d_in[0];   // [N,156,325] fp32 (0/1 spikes)
    const float* W1 = (const float*)d_in[1];   // [32,156]
    const float* W2 = (const float*)d_in[2];   // [20,32]
    float* out = (float*)d_out;                // [N,20,325] fp32 spikes

    const int N = in_sizes[0] / (156 * 325);   // 1024
    slayer_fused<<<dim3(N), dim3(32), 0, stream>>>(X, W1, W2, out);
}
